// Selfatt_26594437496900
// MI455X (gfx1250) — matrix operand order repaired and checked by reading, not compiled, not run on hardware
//
#include <hip/hip_runtime.h>
#include <hip/hip_bf16.h>

typedef __attribute__((ext_vector_type(16))) _Float16 v16h;
typedef __attribute__((ext_vector_type(8)))  _Float16 v8h;
typedef __attribute__((ext_vector_type(4)))  _Float16 v4h;
typedef __attribute__((ext_vector_type(8)))  float    v8f;
typedef __attribute__((ext_vector_type(4)))  float    v4f;
typedef int v4i __attribute__((vector_size(4 * sizeof(int))));  // matches builtin param

#define AS1 __attribute__((address_space(1)))
#define AS3 __attribute__((address_space(3)))
typedef AS3 _Float16 lh_t;   // LDS-resident f16

#define BM   128
#define BN   128
#define BK   32
#define LDK  40    // padded LDS row stride (f16): 80B rows -> conflict-free frag reads
#define LDO  136   // padded LDS out-tile stride
#define SMEM_ELEMS (4 * BM * LDK)   // 20480 f16 = 40960B (>= BM*LDO = 17408 for epilogue)

#if __has_builtin(__builtin_amdgcn_global_load_async_to_lds_b128)
#define USE_ASYNC 1
#else
#define USE_ASYNC 0
#endif

__device__ __forceinline__ void wait_async0() {
#if __has_builtin(__builtin_amdgcn_s_wait_asynccnt)
  __builtin_amdgcn_s_wait_asynccnt(0);
#else
  asm volatile("s_wait_asynccnt 0x0" ::: "memory");
#endif
}

#if USE_ASYNC
// async global->LDS copy of 32B (two b128); INST_OFFSET applies to both addresses
__device__ __forceinline__ void async_cp32(const _Float16* gsrc, lh_t* ldst) {
  __builtin_amdgcn_global_load_async_to_lds_b128((AS1 v4i*)gsrc, (AS3 v4i*)ldst, 0, 0);
  __builtin_amdgcn_global_load_async_to_lds_b128((AS1 v4i*)gsrc, (AS3 v4i*)ldst, 16, 0);
}
#endif

// ---------------------------------------------------------------------------
// fp32 -> f16 conversion (vectorized, grid-stride)
// ---------------------------------------------------------------------------
__global__ __launch_bounds__(256) void cvt_f32_f16(const float* __restrict__ in,
                                                   _Float16* __restrict__ out, int n4) {
  int i = blockIdx.x * blockDim.x + threadIdx.x;
  int stride = gridDim.x * blockDim.x;
  for (; i < n4; i += stride) {
    v4f v = ((const v4f*)in)[i];
    v4h h;
    h[0] = (_Float16)v[0]; h[1] = (_Float16)v[1];
    h[2] = (_Float16)v[2]; h[3] = (_Float16)v[3];
    ((v4h*)out)[i] = h;
  }
}

// ---------------------------------------------------------------------------
// Core 128x128 tile GEMM: C = A[M,K] * Bt[N,K]^T, f16 in, f32 accum.
// Double-buffered LDS fed by async global->LDS copies (ASYNCcnt), one
// workgroup barrier per 32-K step. 8 waves: 4(M) x 2(N), each 32x64.
// All LDS traffic via addrspace(3) pointers -> ds_load/ds_store codegen.
// ---------------------------------------------------------------------------
__device__ __forceinline__ void gemm_tile_f16(const _Float16* __restrict__ A, int lda,
                                              const _Float16* __restrict__ Bt, int ldb,
                                              int K, int m0, int n0,
                                              lh_t* smem, v8f acc[2][4]) {
  const int tid    = threadIdx.x;
  const int lane   = tid & 31;
  const int wave   = tid >> 5;
  const int wm     = wave >> 1;       // 0..3
  const int wn     = wave & 1;        // 0..1
  const int lrow   = tid >> 1;        // 0..127 : staging row
  const int lchunk = (tid & 1) * 16;  // 0 or 16 : staging k-chunk
  const int khalf  = lane >> 4;
  const int frow   = lane & 15;

  lh_t* sA[2] = {smem,                smem + BM * LDK};
  lh_t* sB[2] = {smem + 2 * BM * LDK, smem + 3 * BM * LDK};

  v8f z = {};
#pragma unroll
  for (int mb = 0; mb < 2; ++mb)
#pragma unroll
    for (int nb = 0; nb < 4; ++nb) acc[mb][nb] = z;

  const _Float16* aSrc = A + (size_t)(m0 + lrow) * lda + lchunk;
  const _Float16* bSrc = Bt + (size_t)(n0 + lrow) * ldb + lchunk;
  const int sOff = lrow * LDK + lchunk;
  const int nk = K / BK;

#if USE_ASYNC
  // prologue: tile 0 -> buffer 0
  async_cp32(aSrc, sA[0] + sOff);
  async_cp32(bSrc, sB[0] + sOff);
#endif

  for (int i = 0; i < nk; ++i) {
    const int cur = i & 1;
#if USE_ASYNC
    wait_async0();       // this wave's tile-i loads have landed in LDS
    __syncthreads();     // all waves done reading buffer (i+1)&1 (used at i-1)
    if (i + 1 < nk) {
      const int nxt = cur ^ 1;
      async_cp32(aSrc + (i + 1) * BK, sA[nxt] + sOff);
      async_cp32(bSrc + (i + 1) * BK, sB[nxt] + sOff);
    }
#else
    // synchronous fallback: regs -> LDS, two barriers
    const v8h* ag = (const v8h*)(aSrc + i * BK);
    v8h a0 = ag[0], a1 = ag[1];
    const v8h* bg = (const v8h*)(bSrc + i * BK);
    v8h b0 = bg[0], b1 = bg[1];
    __syncthreads();
    *(AS3 v8h*)&sA[cur][sOff]     = a0;
    *(AS3 v8h*)&sA[cur][sOff + 8] = a1;
    *(AS3 v8h*)&sB[cur][sOff]     = b0;
    *(AS3 v8h*)&sB[cur][sOff + 8] = b1;
    __syncthreads();
#endif

    // A fragments: lanes 0-15 = rows, contiguous 16 K per half-wave
    v16h af[2];
#pragma unroll
    for (int mb = 0; mb < 2; ++mb) {
      const lh_t* p = &sA[cur][(wm * 32 + mb * 16 + frow) * LDK + khalf * 16];
#pragma unroll
      for (int j = 0; j < 8; ++j) { af[mb][j] = p[j]; af[mb][j + 8] = p[8 + j]; }
    }
    // B fragments: lanes 0-15 = cols, contiguous 16 K per half-wave
    v16h bf[4];
#pragma unroll
    for (int nb = 0; nb < 4; ++nb) {
      const lh_t* q = &sB[cur][(wn * 64 + nb * 16 + frow) * LDK + khalf * 16];
#pragma unroll
      for (int j = 0; j < 16; ++j) bf[nb][j] = q[j];
    }

#pragma unroll
    for (int mb = 0; mb < 2; ++mb)
#pragma unroll
      for (int nb = 0; nb < 4; ++nb)
        acc[mb][nb] = __builtin_amdgcn_wmma_f32_16x16x32_f16(
            false, af[mb], false, bf[nb], (short)0, acc[mb][nb], false, false);
  }
}

// stage fp32 accumulators into an LDS f16 tile [BM][LDO]
__device__ __forceinline__ void stage_out_f16(v8f acc[2][4], lh_t* sOut) {
  const int lane = threadIdx.x & 31;
  const int wave = threadIdx.x >> 5;
  const int rowb = (wave >> 1) * 32;
  const int colb = (wave & 1) * 64;
#pragma unroll
  for (int mb = 0; mb < 2; ++mb)
#pragma unroll
    for (int nb = 0; nb < 4; ++nb)
#pragma unroll
      for (int e = 0; e < 8; ++e) {
        int row = rowb + mb * 16 + (lane >> 4) * 8 + e;
        int col = colb + nb * 16 + (lane & 15);
        sOut[row * LDO + col] = (_Float16)acc[mb][nb][e];
      }
}

// coalesced tile store from LDS; TRANS=true writes dst[c*ldd + r]
template <bool TRANS>
__device__ __forceinline__ void store_tile_f16(const lh_t* sOut,
                                               _Float16* __restrict__ dst, int ldd) {
  const int t    = threadIdx.x;
  const int rr   = t >> 1;         // output row (or col if TRANS)
  const int half = (t & 1) * 64;
#pragma unroll
  for (int j = 0; j < 8; ++j) {
    v8h v;
    if (!TRANS) {
      v = *(const AS3 v8h*)&sOut[rr * LDO + half + j * 8];
    } else {
#pragma unroll
      for (int e = 0; e < 8; ++e) v[e] = sOut[(half + j * 8 + e) * LDO + rr];
    }
    *(v8h*)&dst[(size_t)rr * ldd + half + j * 8] = v;
  }
}

// ---------------------------------------------------------------------------
// Kernel 1: fused projections  Y = x_h * W_h^T  ([32768,1024] x [1536,1024]^T)
// cols [0,256)->theta (row-major), [256,512)->phi^T, [512,1536)->g^T
// ---------------------------------------------------------------------------
__global__ __launch_bounds__(256) void proj_kernel(const _Float16* __restrict__ xh,
                                                   const _Float16* __restrict__ wh,
                                                   _Float16* __restrict__ theta,
                                                   _Float16* __restrict__ phiT,
                                                   _Float16* __restrict__ gT) {
  __shared__ __align__(16) _Float16 smem[SMEM_ELEMS];
  lh_t* s3 = (lh_t*)smem;
  v8f acc[2][4];
  const int m0 = blockIdx.y * BM;  // row over B*N = 32768
  const int n0 = blockIdx.x * BN;  // col over 1536
  gemm_tile_f16(xh, 1024, wh, 1024, 1024, m0, n0, s3, acc);
  __syncthreads();                 // buffers free; smem reused as epilogue tile
  stage_out_f16(acc, s3);
  __syncthreads();
  const int b = m0 >> 12, nloc = m0 & 4095;
  if (n0 < 256) {
    _Float16* dst = theta + (size_t)b * 4096 * 256 + (size_t)nloc * 256 + n0;
    store_tile_f16<false>(s3, dst, 256);
  } else if (n0 < 512) {
    _Float16* dst = phiT + (size_t)b * 256 * 4096 + (size_t)(n0 - 256) * 4096 + nloc;
    store_tile_f16<true>(s3, dst, 4096);
  } else {
    _Float16* dst = gT + (size_t)b * 1024 * 4096 + (size_t)(n0 - 512) * 4096 + nloc;
    store_tile_f16<true>(s3, dst, 4096);
  }
}

// ---------------------------------------------------------------------------
// Kernel 2: M = phi^T * g  per batch ([256,4096] x [1024,4096]^T), K=4096
// written transposed as MT[O=1024][L=256]
// ---------------------------------------------------------------------------
__global__ __launch_bounds__(256) void pg_kernel(const _Float16* __restrict__ phiT,
                                                 const _Float16* __restrict__ gT,
                                                 _Float16* __restrict__ MT) {
  __shared__ __align__(16) _Float16 smem[SMEM_ELEMS];
  lh_t* s3 = (lh_t*)smem;
  v8f acc[2][4];
  const int b  = blockIdx.z;
  const int m0 = blockIdx.y * BM;  // over L=256
  const int n0 = blockIdx.x * BN;  // over O=1024
  gemm_tile_f16(phiT + (size_t)b * 256 * 4096, 4096,
                gT   + (size_t)b * 1024 * 4096, 4096, 4096, m0, n0, s3, acc);
  __syncthreads();
  stage_out_f16(acc, s3);
  __syncthreads();
  _Float16* dst = MT + (size_t)b * 1024 * 256 + (size_t)n0 * 256 + m0;
  store_tile_f16<true>(s3, dst, 256);
}

// ---------------------------------------------------------------------------
// Kernel 3: out = scale * theta * MT^T per batch ([4096,256] x [1024,256]^T)
// fp32 direct stores (16 consecutive floats per half-wave -> 64B segments)
// ---------------------------------------------------------------------------
__global__ __launch_bounds__(256) void out_kernel(const _Float16* __restrict__ theta,
                                                  const _Float16* __restrict__ MT,
                                                  float* __restrict__ out, float scale) {
  __shared__ __align__(16) _Float16 smem[SMEM_ELEMS];
  lh_t* s3 = (lh_t*)smem;
  v8f acc[2][4];
  const int b  = blockIdx.z;
  const int m0 = blockIdx.y * BM;  // over N=4096
  const int n0 = blockIdx.x * BN;  // over O=1024
  gemm_tile_f16(theta + (size_t)b * 4096 * 256, 256,
                MT    + (size_t)b * 1024 * 256, 256, 256, m0, n0, s3, acc);
  const int lane = threadIdx.x & 31;
  const int wave = threadIdx.x >> 5;
  const int rowb = m0 + (wave >> 1) * 32;
  const int colb = n0 + (wave & 1) * 64;
  float* dst = out + (size_t)b * 4096 * 1024;
#pragma unroll
  for (int mb = 0; mb < 2; ++mb)
#pragma unroll
    for (int nb = 0; nb < 4; ++nb)
#pragma unroll
      for (int e = 0; e < 8; ++e) {
        int row = rowb + mb * 16 + (lane >> 4) * 8 + e;
        int col = colb + nb * 16 + (lane & 15);
        dst[(size_t)row * 1024 + col] = acc[mb][nb][e] * scale;
      }
}

// ---------------------------------------------------------------------------
extern "C" void kernel_launch(void* const* d_in, const int* in_sizes, int n_in,
                              void* d_out, int out_size, void* d_ws, size_t ws_size,
                              hipStream_t stream) {
  const float* x  = (const float*)d_in[0];  // [8,4096,1024]
  const float* Wt = (const float*)d_in[1];  // [256,1024]
  const float* Wp = (const float*)d_in[2];  // [256,1024]
  const float* Wg = (const float*)d_in[3];  // [1024,1024]
  float* out = (float*)d_out;               // [8,4096,1024]

  // workspace layout (f16 elements), ~167 MB total
  _Float16* xh = (_Float16*)d_ws;           // 32768 x 1024
  _Float16* wh = xh + (size_t)32768 * 1024; // 1536 x 1024 (theta|phi|g rows)
  _Float16* th = wh + (size_t)1536 * 1024;  // theta  [8][4096][256]
  _Float16* ph = th + (size_t)8 * 4096 * 256;   // phi^T [8][256][4096]
  _Float16* gh = ph + (size_t)8 * 256 * 4096;   // g^T   [8][1024][4096]
  _Float16* mh = gh + (size_t)8 * 1024 * 4096;  // M^T   [8][1024][256]

  // fp32 -> f16 conversions
  cvt_f32_f16<<<2048, 256, 0, stream>>>(x, xh, (32768 * 1024) / 4);
  cvt_f32_f16<<<64, 256, 0, stream>>>(Wt, wh, (256 * 1024) / 4);
  cvt_f32_f16<<<64, 256, 0, stream>>>(Wp, wh + 256 * 1024, (256 * 1024) / 4);
  cvt_f32_f16<<<256, 256, 0, stream>>>(Wg, wh + 512 * 1024, (1024 * 1024) / 4);

  // 1) projections: [32768,1536] = x_h * W_h^T
  dim3 g1(12, 256, 1);
  proj_kernel<<<g1, 256, 0, stream>>>(xh, wh, th, ph, gh);

  // 2) M = phi^T * g per batch (K=4096)
  dim3 g2(8, 2, 8);
  pg_kernel<<<g2, 256, 0, stream>>>(ph, gh, mh);

  // 3) out = scale * theta * M  (K=256), scale = 1/sqrt(1024)
  dim3 g3(8, 32, 8);
  out_kernel<<<g3, 256, 0, stream>>>(th, mh, out, 1.0f / 32.0f);
}
